// SelfAttention_11785390260643
// MI455X (gfx1250) — matrix operand order repaired and checked by reading, not compiled, not run on hardware
//
#include <hip/hip_runtime.h>
#include <cstdint>
#include <cstddef>

// ---------------------------------------------------------------------------
// Self-attention forward for MI455X (gfx1250, wave32, WMMA).
//   x:     [8, 2048, 512]  f32
//   w_qkv: [512, 1536]     f32
//   out:   [8, 2048, 512]  f32
// All GEMMs run on v_wmma_f32_16x16x32_bf16 (bf16 operands, f32 accum).
// ---------------------------------------------------------------------------

typedef __attribute__((ext_vector_type(16))) __bf16       v16bf;
typedef __attribute__((ext_vector_type(8)))  float        v8f;
typedef __attribute__((ext_vector_type(4)))  unsigned int u32x4;
typedef __attribute__((ext_vector_type(4)))  int          v4i;
typedef __attribute__((ext_vector_type(4)))  float        f32x4;

union BV { v16bf v; u32x4 q[2]; };

#define BATCH 8
#define SEQ   2048
#define DIM   512
#define E3    1536
#define ATT_SCALE 0.044194173824159216f   // 512^-0.5

// LDS row padding to kill bank conflicts (64 banks x 4B):
#define S_STRIDE 65   // floats: (65*row + col) % 64 == (row+col) % 64 -> conflict-free
#define P_STRIDE 72   // bf16:   144B/row keeps 16B alignment, 16-bank spread

#if __has_builtin(__builtin_amdgcn_global_load_async_to_lds_b128) && \
    __has_builtin(__builtin_amdgcn_s_wait_asynccnt)
#define HAVE_ASYNC_LDS 1
typedef __attribute__((address_space(1))) v4i* gv4i_p;
typedef __attribute__((address_space(3))) v4i* sv4i_p;
#endif

static __device__ __forceinline__ v8f wmma_bf16(v16bf a, v16bf b, v8f c) {
    // (neg_a, A, neg_b, B, c_mod, C, reuse_a, reuse_b)
    return __builtin_amdgcn_wmma_f32_16x16x32_bf16(false, a, false, b, (short)0, c,
                                                   false, false);
}

// ---------------------------------------------------------------------------
// Kernel 0: W [512][1536] f32  ->  Wt [1536][512] bf16 (transposed, row-major
// in K so B-fragments of the QKV GEMM are contiguous 16B loads).
// ---------------------------------------------------------------------------
__global__ void wt_cvt_kernel(const float* __restrict__ w, __bf16* __restrict__ wt) {
    int idx = blockIdx.x * blockDim.x + threadIdx.x;
    if (idx >= E3 * DIM) return;
    int e = idx / DIM;
    int d = idx - e * DIM;
    wt[idx] = (__bf16)w[(size_t)d * E3 + e];
}

// ---------------------------------------------------------------------------
// Kernel 1: QKV GEMM.  C[16384,1536] = x[16384,512] @ W[512,1536].
// One wave computes a 16x64 tile (A fragment reused across 4 B tiles).
// Q,K stored row-major bf16 [b][n][512]; V stored transposed bf16 [b][512][n].
// ---------------------------------------------------------------------------
__global__ void __launch_bounds__(256) qkv_gemm_kernel(
    const float*  __restrict__ x,
    const __bf16* __restrict__ wt,
    __bf16* __restrict__ qws,
    __bf16* __restrict__ kws,
    __bf16* __restrict__ vtws)
{
    const int lane = threadIdx.x & 31;
    // Wave index and everything derived from it is wave-uniform; force it into
    // an SGPR so the Q/K/V epilogue branch is scalar (s_cbranch, no exec mask).
    const int wave   = __builtin_amdgcn_readfirstlane(threadIdx.x >> 5);
    const int gw     = blockIdx.x * 8 + wave;
    const int tile_m = gw / 24;            // 16384/16 = 1024 row tiles
    const int strip  = gw - tile_m * 24;   // 1536/64  = 24 column strips
    const int m0   = tile_m * 16;
    const int n0   = strip * 64;
    const int half = lane >> 4;
    const int l16  = lane & 15;

    v8f acc[4] = {};

    const float* ap = x + (size_t)(m0 + l16) * DIM;   // A row for this lane

    for (int k0 = 0; k0 < DIM; k0 += 32) {
        // A fragment 16x32: two contiguous 8-float runs per lane -> bf16
        f32x4 a0 = *(const f32x4*)(ap + k0 + half * 8);
        f32x4 a1 = *(const f32x4*)(ap + k0 + half * 8 + 4);
        f32x4 a2 = *(const f32x4*)(ap + k0 + 16 + half * 8);
        f32x4 a3 = *(const f32x4*)(ap + k0 + 16 + half * 8 + 4);
        v16bf A;
        #pragma unroll
        for (int i = 0; i < 4; ++i) {
            A[i]      = (__bf16)a0[i];
            A[i + 4]  = (__bf16)a1[i];
            A[i + 8]  = (__bf16)a2[i];
            A[i + 12] = (__bf16)a3[i];
        }
        // 4 B fragments 32x16, two 8-bf16 runs per lane from transposed weights
        #pragma unroll
        for (int j = 0; j < 4; ++j) {
            const __bf16* bp = wt + (size_t)(n0 + j * 16 + l16) * DIM + k0 + half * 8;
            BV B;
            B.q[0] = *(const u32x4*)(bp);
            B.q[1] = *(const u32x4*)(bp + 16);
            acc[j] = wmma_bf16(A, B.v, acc[j]);
        }
    }

    // Store: C/D layout -> m = r + 8*half, n = l16.  strip is in an SGPR, so
    // this three-way split compiles to scalar branches.
    const int rbase = m0 + half * 8;
    if (strip < 8) {                                   // Q columns 0..511
        #pragma unroll
        for (int j = 0; j < 4; ++j) {
            const int e = n0 + j * 16 + l16;
            #pragma unroll
            for (int r = 0; r < 8; ++r)
                qws[(size_t)(rbase + r) * DIM + e] = (__bf16)acc[j][r];
        }
    } else if (strip < 16) {                           // K columns 512..1023
        #pragma unroll
        for (int j = 0; j < 4; ++j) {
            const int e = n0 + j * 16 + l16 - 512;
            #pragma unroll
            for (int r = 0; r < 8; ++r)
                kws[(size_t)(rbase + r) * DIM + e] = (__bf16)acc[j][r];
        }
    } else {                                           // V columns, transposed
        const int bb = rbase >> 11;                    // tile never crosses batch
        #pragma unroll
        for (int j = 0; j < 4; ++j) {
            const int d = n0 + j * 16 + l16 - 1024;
            #pragma unroll
            for (int r = 0; r < 8; ++r)
                vtws[((size_t)bb * DIM + d) * SEQ + ((rbase + r) & 2047)] =
                    (__bf16)acc[j][r];
        }
    }
}

// ---------------------------------------------------------------------------
// Kernel 2: flash attention.  One block (8 wave32) per (batch, 32-query block).
// Per 64-key step: 8 waves each do one 16x16 S tile (16 WMMAs over d=512),
// wave0 does the online softmax (one lane per query row), then every wave
// accumulates its private 32q x 64d O slice with 16 WMMAs (P bf16 from LDS,
// V^T contiguous from global).
// ---------------------------------------------------------------------------
__global__ void __launch_bounds__(256) attn_kernel(
    const __bf16* __restrict__ qws,
    const __bf16* __restrict__ kws,
    const __bf16* __restrict__ vtws,
    float* __restrict__ out)
{
    __shared__ __bf16 Qs[32 * DIM];           // 32 KB of the 320 KB WGP LDS
    __shared__ float  Sbuf[32 * S_STRIDE];    // padded: conflict-free softmax
    __shared__ __bf16 Pbuf[32 * P_STRIDE];    // padded: 16B-aligned fragments
    __shared__ float  row_max[32], row_sum[32], row_alpha[32];

    const int b    = blockIdx.x >> 6;
    const int q0   = (blockIdx.x & 63) * 32;
    const int lane = threadIdx.x & 31;
    // Scalarize the wave index -> qi/kj/d0 become SGPRs and global base
    // addresses can use the saddr+voffset form.
    const int wave = __builtin_amdgcn_readfirstlane(threadIdx.x >> 5);
    const int half = lane >> 4;
    const int l16  = lane & 15;
    const int qi   = wave >> 2;            // S-tile query sub-block (0..1)
    const int kj   = wave & 3;             // S-tile key   sub-block (0..3)
    const int d0   = wave * 64;            // this wave's O d-block

    // Stage the 32x512 bf16 Q block into LDS: async DMA path if available.
    {
        const __bf16* src = qws + ((size_t)b * SEQ + q0) * DIM;
#ifdef HAVE_ASYNC_LDS
        for (int i = threadIdx.x; i < (32 * DIM) / 8; i += 256) {
            __builtin_amdgcn_global_load_async_to_lds_b128(
                (gv4i_p)(src + i * 8),
                (sv4i_p)(Qs + i * 8),
                0, 0);
        }
        __builtin_amdgcn_s_wait_asynccnt(0);
#else
        const u32x4* s4 = (const u32x4*)src;
        u32x4* dst = (u32x4*)Qs;
        for (int i = threadIdx.x; i < (32 * DIM * 2) / 16; i += 256) dst[i] = s4[i];
#endif
    }
    if (threadIdx.x < 32) { row_max[threadIdx.x] = -3.0e38f; row_sum[threadIdx.x] = 0.0f; }
    __syncthreads();

    v8f o[8] = {};                         // 2 q-subtiles x 4 d-subtiles

    for (int ks = 0; ks < SEQ; ks += 64) {
        // ---- S = Q @ K^T for this wave's 16x16 tile, K-dim = 512
        v8f s = {};
        const __bf16* qrow = Qs + (qi * 16 + l16) * DIM;
        const __bf16* krow = kws + ((size_t)b * SEQ + ks + kj * 16 + l16) * DIM;
        if (ks + 64 < SEQ) {
            __builtin_prefetch((const void*)(krow + 64 * DIM), 0, 0);
            __builtin_prefetch((const void*)(vtws + ((size_t)b * DIM + d0 + l16) * SEQ
                                             + ks + 64), 0, 0);
        }
        #pragma unroll 4
        for (int dk = 0; dk < DIM; dk += 32) {
            BV A, B;
            A.q[0] = *(const u32x4*)(qrow + dk + half * 8);
            A.q[1] = *(const u32x4*)(qrow + dk + 16 + half * 8);
            B.q[0] = *(const u32x4*)(krow + dk + half * 8);
            B.q[1] = *(const u32x4*)(krow + dk + 16 + half * 8);
            s = wmma_bf16(A.v, B.v, s);
        }
        #pragma unroll
        for (int r = 0; r < 8; ++r)
            Sbuf[(qi * 16 + half * 8 + r) * S_STRIDE + kj * 16 + l16] = s[r] * ATT_SCALE;
        __syncthreads();

        // ---- online softmax: one lane per query row (conflict-free strides)
        if (threadIdx.x < 32) {
            const int row = threadIdx.x;
            float mold = row_max[row];
            float m = mold;
            #pragma unroll 8
            for (int j = 0; j < 64; ++j) m = fmaxf(m, Sbuf[row * S_STRIDE + j]);
            float alpha = __expf(mold - m);
            float sum   = row_sum[row] * alpha;
            #pragma unroll 8
            for (int j = 0; j < 64; ++j) {
                float p = __expf(Sbuf[row * S_STRIDE + j] - m);
                sum += p;
                Pbuf[row * P_STRIDE + j] = (__bf16)p;
            }
            row_max[row]   = m;
            row_sum[row]   = sum;
            row_alpha[row] = alpha;
        }
        __syncthreads();

        // ---- O += P @ V on this wave's 32q x 64d slice (rescale by alpha)
        #pragma unroll
        for (int qs = 0; qs < 2; ++qs) {
            float al[8];
            #pragma unroll
            for (int r = 0; r < 8; ++r) al[r] = row_alpha[qs * 16 + half * 8 + r];
            #pragma unroll
            for (int dj = 0; dj < 4; ++dj) {
                v8f acc = o[qs * 4 + dj];
                #pragma unroll
                for (int r = 0; r < 8; ++r) acc[r] *= al[r];
                #pragma unroll
                for (int c = 0; c < 2; ++c) {   // two K=32 chunks of 64 keys
                    const __bf16* pp = Pbuf + (qs * 16 + l16) * P_STRIDE + c * 32;
                    BV A, B;
                    A.q[0] = *(const u32x4*)(pp + half * 8);
                    A.q[1] = *(const u32x4*)(pp + 16 + half * 8);
                    const __bf16* vp = vtws +
                        ((size_t)b * DIM + d0 + dj * 16 + l16) * SEQ + ks + c * 32 + half * 8;
                    B.q[0] = *(const u32x4*)(vp);
                    B.q[1] = *(const u32x4*)(vp + 16);
                    acc = wmma_bf16(A.v, B.v, acc);
                }
                o[qs * 4 + dj] = acc;
            }
        }
        __syncthreads();
    }

    // ---- finalize: divide by softmax denominator, write f32 output
    #pragma unroll
    for (int qs = 0; qs < 2; ++qs) {
        float inv[8];
        #pragma unroll
        for (int r = 0; r < 8; ++r) inv[r] = 1.0f / row_sum[qs * 16 + half * 8 + r];
        #pragma unroll
        for (int dj = 0; dj < 4; ++dj) {
            #pragma unroll
            for (int r = 0; r < 8; ++r) {
                out[((size_t)b * SEQ + q0 + qs * 16 + half * 8 + r) * DIM +
                    d0 + dj * 16 + l16] = o[qs * 4 + dj][r] * inv[r];
            }
        }
    }
}

// ---------------------------------------------------------------------------
extern "C" void kernel_launch(void* const* d_in, const int* in_sizes, int n_in,
                              void* d_out, int out_size, void* d_ws, size_t ws_size,
                              hipStream_t stream) {
    (void)in_sizes; (void)n_in; (void)out_size; (void)ws_size;
    const float* x = (const float*)d_in[0];
    const float* w = (const float*)d_in[1];
    float* out = (float*)d_out;

    char* ws = (char*)d_ws;
    const size_t WT_BYTES  = (size_t)E3 * DIM * 2;            // 1.5 MB
    const size_t QKV_BYTES = (size_t)BATCH * SEQ * DIM * 2;   // 16 MB each
    __bf16* wt   = (__bf16*)(ws);
    __bf16* qws  = (__bf16*)(ws + WT_BYTES);
    __bf16* kws  = (__bf16*)(ws + WT_BYTES + QKV_BYTES);
    __bf16* vtws = (__bf16*)(ws + WT_BYTES + 2 * QKV_BYTES);

    wt_cvt_kernel<<<(E3 * DIM + 255) / 256, 256, 0, stream>>>(w, wt);

    // 1024 row tiles * 24 strips = 24576 waves, 8 waves/block
    qkv_gemm_kernel<<<(BATCH * SEQ / 16) * (E3 / 64) / 8, 256, 0, stream>>>(
        x, wt, qws, kws, vtws);

    // 8 batches * 64 query blocks
    attn_kernel<<<BATCH * (SEQ / 32), 256, 0, stream>>>(qws, kws, vtws, out);
}
